// IsoMaxPlusLossFirstPart_75780402971149
// MI455X (gfx1250) — matrix operand order repaired and checked by reading, not compiled, not run on hardware
//
#include <hip/hip_runtime.h>
#include <hip/hip_bf16.h>

typedef __attribute__((ext_vector_type(16))) _Float16 v16h;
typedef __attribute__((ext_vector_type(8)))  _Float16 v8h;
typedef __attribute__((ext_vector_type(8)))  float    v8f;

#define B_   16
#define C_   256
#define H_   128
#define W_   256
#define K_   19
#define HW_  (H_ * W_)          // 32768
#define KPAD 32
#define TILES_PER_WAVE  8
#define WAVES_PER_BLOCK 8

__global__ __launch_bounds__(256)
void isomax_logits_kernel(const float* __restrict__ feat,
                          const float* __restrict__ protos,
                          const float* __restrict__ dscale,
                          float* __restrict__ out)
{
    // Normalized prototypes in f16, padded to 32 rows: 32*256*2 = 16 KB LDS
    __shared__ _Float16 sP[KPAD * C_];
    __shared__ float sRed[K_ * 8];
    __shared__ float sRn[K_];

    const int tid = threadIdx.x;

    // ---- Phase 1: L2-normalize prototypes into LDS as f16 ----
    if (tid < K_ * 8) {
        const int p = tid >> 3, s = tid & 7;
        const float* row = protos + p * C_ + s * 32;
        float acc = 0.f;
        #pragma unroll
        for (int i = 0; i < 32; ++i) acc += row[i] * row[i];
        sRed[tid] = acc;
    }
    __syncthreads();
    if (tid < K_) {
        float acc = 0.f;
        #pragma unroll
        for (int s = 0; s < 8; ++s) acc += sRed[tid * 8 + s];
        sRn[tid] = 1.0f / fmaxf(sqrtf(acc), 1e-12f);
    }
    __syncthreads();
    for (int idx = tid; idx < KPAD * C_; idx += 256) {
        const int p = idx >> 8, c = idx & (C_ - 1);
        const float v = (p < K_) ? protos[p * C_ + c] * sRn[p] : 0.0f;
        sP[idx] = (_Float16)v;
    }
    __syncthreads();

    const float scale = fabsf(dscale[0]);

    const int lane = tid & 31;
    const int wave = tid >> 5;
    const int n    = lane & 15;   // pixel-in-tile == WMMA column
    const int hi   = lane >> 4;   // half-wave selector

    // Each wave owns 128 consecutive pixels (half of one (b,h) row of W=256)
    const int gwave = blockIdx.x * WAVES_PER_BLOCK + wave;
    const int pix0  = gwave * (TILES_PER_WAVE * 16);
    const int b     = pix0 >> 15;            // / HW_
    const int rem   = pix0 & (HW_ - 1);
    const int h     = rem >> 8;              // / W_
    const int wbase = rem & (W_ - 1);

    const float* featRow = feat + (size_t)b * C_ * HW_ + (size_t)h * W_;
    float*       outRow  = out  + (size_t)b * K_ * HW_ + (size_t)h * W_;

    for (int t = 0; t < TILES_PER_WAVE; ++t) {
        const int w0 = wbase + t * 16;
        const float* fp = featRow + w0 + n;   // + ch*HW_ per channel

        v8f c0 = {};   // protos 0..15
        v8f c1 = {};   // protos 16..31 (only 16..18 stored)
        float ssq = 0.f;

        #pragma unroll
        for (int kc = 0; kc < C_ / 32; ++kc) {
            // --- B fragment: 32 channels x 16 pixels (global f32 -> f16)
            // lanes 0-15: K = 0..15 of chunk, lanes 16-31: K = 16..31
            const int chBase = kc * 32 + hi * 16;
            v16h bv;
            #pragma unroll
            for (int e = 0; e < 16; ++e) {
                const float v = fp[(size_t)(chBase + e) * HW_];
                ssq += v * v;
                bv[e] = (_Float16)v;
            }

            // --- A fragments from LDS (proto rows), in a contiguous (not table-order)
            // 16x32 layout: lane m holds K = hi*16+{0..15}
            const int abase0 = n * C_        + kc * 32 + hi * 16;
            const int abase1 = (16 + n) * C_ + kc * 32 + hi * 16;
            const v8h a0lo = *(const v8h*)(sP + abase0);
            const v8h a0hi = *(const v8h*)(sP + abase0 + 8);
            const v8h a1lo = *(const v8h*)(sP + abase1);
            const v8h a1hi = *(const v8h*)(sP + abase1 + 8);
            v16h a0, a1;
            #pragma unroll
            for (int i = 0; i < 8; ++i) {
                a0[i] = a0lo[i]; a0[8 + i] = a0hi[i];
                a1[i] = a1lo[i]; a1[8 + i] = a1hi[i];
            }

            c0 = __builtin_amdgcn_wmma_f32_16x16x32_f16(
                     false, a0, false, bv, (short)0, c0, false, false);
            c1 = __builtin_amdgcn_wmma_f32_16x16x32_f16(
                     false, a1, false, bv, (short)0, c1, false, false);
        }

        // Per-pixel squared norm: lane n has half the channels, lane n+16 the rest
        ssq += __shfl_xor(ssq, 16, 32);
        const float rn = 1.0f / fmaxf(sqrtf(ssq), 1e-12f);

        // Epilogue: d = sqrt(max(2 - 2*cos, 0)), coalesced stores over lanes
        #pragma unroll
        for (int r = 0; r < 8; ++r) {
            const int k = r + hi * 8;                 // proto row from C layout
            const float cosv = c0[r] * rn;
            const float d = sqrtf(fmaxf(2.f - 2.f * cosv, 0.f));
            outRow[(size_t)k * HW_ + w0 + n] = -scale * d;
        }
        #pragma unroll
        for (int r = 0; r < 3; ++r) {
            const int k = 16 + r + hi * 8;            // 16..18 valid (hi==0)
            if (k < K_) {
                const float cosv = c1[r] * rn;
                const float d = sqrtf(fmaxf(2.f - 2.f * cosv, 0.f));
                outRow[(size_t)k * HW_ + w0 + n] = -scale * d;
            }
        }
    }
}

extern "C" void kernel_launch(void* const* d_in, const int* in_sizes, int n_in,
                              void* d_out, int out_size, void* d_ws, size_t ws_size,
                              hipStream_t stream) {
    const float* feat   = (const float*)d_in[0];
    const float* protos = (const float*)d_in[1];
    const float* dscale = (const float*)d_in[2];
    float* out = (float*)d_out;

    const int N      = in_sizes[0] / C_;                       // 524288 pixels
    const int blocks = N / (16 * TILES_PER_WAVE * WAVES_PER_BLOCK); // 512

    hipLaunchKernelGGL(isomax_logits_kernel, dim3(blocks), dim3(256), 0, stream,
                       feat, protos, dscale, out);
}
